// Net_67723044323452
// MI455X (gfx1250) — compile-verified, matrix operand order checked by reading, not run on hardware
//
#include <hip/hip_runtime.h>
#include <hip/hip_bf16.h>
#include <math.h>

// ---------------- types ----------------
typedef __bf16 bf16;
typedef __attribute__((ext_vector_type(16))) __bf16 v16bf;
typedef __attribute__((ext_vector_type(8)))  float  v8f;

// ---------------- f32 -> bf16 conversion (vectorized) ----------------
__global__ __launch_bounds__(256) void cvt_kernel(const float* __restrict__ a,
                                                  bf16* __restrict__ o, long long n) {
    long long i = ((long long)blockIdx.x * 256 + threadIdx.x) * 4;
    if (i + 3 < n) {
        float4 v = *reinterpret_cast<const float4*>(a + i);
        union { uint2 u; bf16 h[4]; } r;
        r.h[0] = (bf16)v.x; r.h[1] = (bf16)v.y; r.h[2] = (bf16)v.z; r.h[3] = (bf16)v.w;
        *reinterpret_cast<uint2*>(o + i) = r.u;
    } else {
        for (long long j = i; j < n; ++j) o[j] = (bf16)a[j];
    }
}

// ---------------- token embedding + sinusoidal PE ----------------
__global__ __launch_bounds__(256) void embed_kernel(const float* __restrict__ tok_emb,
                                                    const int* __restrict__ ids,
                                                    float* __restrict__ xf,
                                                    bf16* __restrict__ xb,
                                                    int NB, float scale) {
    long long row = blockIdx.x;
    int l = (int)(row / NB);
    int tok = ids[row];
    int t = threadIdx.x;
    for (int i = 0; i < 2; ++i) {
        int d = t + i * 256;
        int half = d >> 1;
        float ang = (float)l * __powf(10000.0f, -2.0f * (float)half / 512.0f);
        float pe = (d & 1) ? __cosf(ang) : __sinf(ang);
        float v = tok_emb[(long long)tok * 512 + d] * scale + pe;
        xf[row * 512 + d] = v;
        xb[row * 512 + d] = (bf16)v;
    }
}

// ---------------- generic batched bf16 WMMA GEMM ----------------
// C[z] = A[z] * op(B[z]) (+bias) (+relu).  A row-major [M,K].
// TRANSB=1: B row-major [N,K];  TRANSB=0: B row-major [K,N].
// Batch: z1=z/nz2, z2=z%nz2, offset = z1*s1 + z2*s2 per matrix.
// ALIGNED=1 (M%128==0, N%64==0, K%32==0): stage tiles with
// global_load_async_to_lds_b64 (ASYNCcnt), double-buffered ping-pong,
// and unguarded epilogue.
#define TM 128
#define TN 64
#define TK 32
#define LDS_STRIDE 36

template <int ALIGNED, int TRANSB>
__global__ __launch_bounds__(256) void gemm_bf16_kernel(
    const bf16* __restrict__ A, long long aS1, long long aS2, int lda,
    const bf16* __restrict__ B, long long bS1, long long bS2, int ldb,
    float* __restrict__ Cf, bf16* __restrict__ Cb, long long cS1, long long cS2, int ldc,
    const float* __restrict__ bias, int relu,
    int M, int N, int K, int nz2)
{
    __shared__ bf16 sA[2][TM * LDS_STRIDE];
    __shared__ bf16 sB[2][TN * LDS_STRIDE];

    int z  = blockIdx.z;
    int z1 = z / nz2, z2 = z % nz2;
    const bf16* Ab = A + (long long)z1 * aS1 + (long long)z2 * aS2;
    const bf16* Bb = B + (long long)z1 * bS1 + (long long)z2 * bS2;
    long long cOff = (long long)z1 * cS1 + (long long)z2 * cS2;

    int rowBase = blockIdx.y * TM;
    int colBase = blockIdx.x * TN;
    int t = threadIdx.x;
    int wave = t >> 5, lane = t & 31;
    int wm = wave >> 1, wn = wave & 1;          // 4x2 wave grid, 32x32 out per wave

    auto stage = [&](int k0, int buf) {
        // ---- A tile 128x32 ----
        #pragma unroll
        for (int i = 0; i < 4; ++i) {
            int c  = t + i * 256;               // chunk of 4 bf16 (8B)
            int r  = c >> 3;
            int kc = (c & 7) * 4;
            int gr = rowBase + r;
            if constexpr (ALIGNED) {
                unsigned ldsa = (unsigned)(size_t)&sA[buf][r * LDS_STRIDE + kc];
                const bf16* src = Ab + (long long)gr * lda + k0 + kc;
                asm volatile("global_load_async_to_lds_b64 %0, %1, off"
                             :: "v"(ldsa), "v"((unsigned long long)(size_t)src)
                             : "memory");
            } else {
                int cr = gr < M ? gr : M - 1;   // clamped, branchless
                uint2 val = *reinterpret_cast<const uint2*>(Ab + (long long)cr * lda + k0 + kc);
                if (gr >= M) { val.x = 0u; val.y = 0u; }
                *reinterpret_cast<uint2*>(&sA[buf][r * LDS_STRIDE + kc]) = val;
            }
        }
        // ---- B tile -> sB[n][k] (64x32) ----
        if constexpr (TRANSB) {
            #pragma unroll
            for (int i = 0; i < 2; ++i) {
                int c  = t + i * 256;
                int n  = c >> 3;
                int kc = (c & 7) * 4;
                int gn = colBase + n;
                if constexpr (ALIGNED) {
                    unsigned ldsb = (unsigned)(size_t)&sB[buf][n * LDS_STRIDE + kc];
                    const bf16* src = Bb + (long long)gn * ldb + k0 + kc;
                    asm volatile("global_load_async_to_lds_b64 %0, %1, off"
                                 :: "v"(ldsb), "v"((unsigned long long)(size_t)src)
                                 : "memory");
                } else {
                    int cn = gn < N ? gn : N - 1;
                    uint2 val = *reinterpret_cast<const uint2*>(Bb + (long long)cn * ldb + k0 + kc);
                    if (gn >= N) { val.x = 0u; val.y = 0u; }
                    *reinterpret_cast<uint2*>(&sB[buf][n * LDS_STRIDE + kc]) = val;
                }
            }
        } else {
            // B[k][n] -> transpose into sB[n][k]; AV-GEMM dims are always full tiles
            #pragma unroll
            for (int i = 0; i < 2; ++i) {
                int e  = t + i * 256;
                int kk = e >> 4;
                int n4 = (e & 15) * 4;
                union { uint2 u; bf16 h[4]; } val;
                val.u = *reinterpret_cast<const uint2*>(
                            Bb + (long long)(k0 + kk) * ldb + colBase + n4);
                #pragma unroll
                for (int j = 0; j < 4; ++j)
                    sB[buf][(n4 + j) * LDS_STRIDE + kk] = val.h[j];
            }
        }
    };

    v8f acc[2][2] = {};
    int nk = (K + TK - 1) / TK;

    stage(0, 0);
    if constexpr (ALIGNED) asm volatile("s_wait_asynccnt 0x0" ::: "memory");
    __syncthreads();

    for (int ki = 0; ki < nk; ++ki) {
        int buf = ki & 1;
        if (ki + 1 < nk) stage((ki + 1) * TK, buf ^ 1);   // prefetch next slab

        // ---- fragments per CDNA5 16-bit 16x32 layout ----
        int mfl = lane & 15, g = lane >> 4;
        union { v16bf v; unsigned u[8]; } afrag[2], bfrag[2];
        #pragma unroll
        for (int s = 0; s < 2; ++s) {
            int ar = wm * 32 + s * 16 + mfl;
            int br = wn * 32 + s * 16 + mfl;
            #pragma unroll
            for (int j = 0; j < 8; ++j) {
                int kb = ((j >> 2) << 4) + (g << 3) + ((j & 3) << 1);
                afrag[s].u[j] = *reinterpret_cast<const unsigned*>(&sA[buf][ar * LDS_STRIDE + kb]);
                bfrag[s].u[j] = *reinterpret_cast<const unsigned*>(&sB[buf][br * LDS_STRIDE + kb]);
            }
        }
        #pragma unroll
        for (int i = 0; i < 2; ++i)
            #pragma unroll
            for (int j = 0; j < 2; ++j)
                acc[i][j] = __builtin_amdgcn_wmma_f32_16x16x32_bf16(
                    false, afrag[i].v, false, bfrag[j].v,
                    (short)0, acc[i][j], false, false);

        if (ki + 1 < nk) {
            if constexpr (ALIGNED) asm volatile("s_wait_asynccnt 0x0" ::: "memory");
            __syncthreads();
        }
    }

    // ---- epilogue: C layout m = v + 8*(lane/16), n = lane%16 ----
    int nloc = lane & 15, gc = lane >> 4;

    // hoisted per-lane bias (2 columns per lane), loaded once
    float bv[2] = {0.0f, 0.0f};
    if (bias) {
        #pragma unroll
        for (int j = 0; j < 2; ++j) {
            int col = colBase + wn * 32 + j * 16 + nloc;
            if constexpr (ALIGNED) bv[j] = bias[col];
            else                   bv[j] = (col < N) ? bias[col] : 0.0f;
        }
    }
    // branchless bias + relu into a local array
    float val[2][2][8];
    #pragma unroll
    for (int i = 0; i < 2; ++i)
        #pragma unroll
        for (int j = 0; j < 2; ++j)
            #pragma unroll
            for (int v = 0; v < 8; ++v) {
                float x = acc[i][j][v] + bv[j];
                val[i][j][v] = relu ? fmaxf(x, 0.0f) : x;
            }

    if constexpr (ALIGNED) {
        // full tile: no guards; uniform branches around whole store loops
        if (Cf) {
            #pragma unroll
            for (int i = 0; i < 2; ++i)
                #pragma unroll
                for (int j = 0; j < 2; ++j)
                    #pragma unroll
                    for (int v = 0; v < 8; ++v) {
                        int row = rowBase + wm * 32 + i * 16 + gc * 8 + v;
                        int col = colBase + wn * 32 + j * 16 + nloc;
                        Cf[cOff + (long long)row * ldc + col] = val[i][j][v];
                    }
        }
        if (Cb) {
            #pragma unroll
            for (int i = 0; i < 2; ++i)
                #pragma unroll
                for (int j = 0; j < 2; ++j)
                    #pragma unroll
                    for (int v = 0; v < 8; ++v) {
                        int row = rowBase + wm * 32 + i * 16 + gc * 8 + v;
                        int col = colBase + wn * 32 + j * 16 + nloc;
                        Cb[cOff + (long long)row * ldc + col] = (bf16)val[i][j][v];
                    }
        }
    } else {
        #pragma unroll
        for (int i = 0; i < 2; ++i)
            #pragma unroll
            for (int j = 0; j < 2; ++j)
                #pragma unroll
                for (int v = 0; v < 8; ++v) {
                    int row = rowBase + wm * 32 + i * 16 + gc * 8 + v;
                    int col = colBase + wn * 32 + j * 16 + nloc;
                    if (row < M && col < N) {
                        long long off = cOff + (long long)row * ldc + col;
                        if (Cf) Cf[off] = val[i][j][v];
                        if (Cb) Cb[off] = (bf16)val[i][j][v];
                    }
                }
    }
}

// ---------------- row softmax with fused scale + causal mask, bf16 out ----------------
__global__ __launch_bounds__(256) void softmax_kernel(const float* __restrict__ S,
                                                      bf16* __restrict__ P,
                                                      int Lq, int Lk, float scale, int causal) {
    long long row = blockIdx.x;                 // z*Lq + l
    int l = (int)(row % Lq);
    const float* s = S + row * (long long)Lk;
    bf16* p = P + row * (long long)Lk;
    __shared__ float red[8];
    int t = threadIdx.x;

    float mx = -3.4e38f;
    for (int m = t; m < Lk; m += 256) {
        float v = s[m] * scale + ((causal && m > l) ? -1e9f : 0.0f);
        mx = fmaxf(mx, v);
    }
    for (int o = 16; o; o >>= 1) mx = fmaxf(mx, __shfl_xor(mx, o, 32));
    if ((t & 31) == 0) red[t >> 5] = mx;
    __syncthreads();
    mx = red[0];
    for (int i = 1; i < 8; ++i) mx = fmaxf(mx, red[i]);
    __syncthreads();

    float sum = 0.0f;
    for (int m = t; m < Lk; m += 256) {
        float v = s[m] * scale + ((causal && m > l) ? -1e9f : 0.0f);
        sum += __expf(v - mx);
    }
    for (int o = 16; o; o >>= 1) sum += __shfl_xor(sum, o, 32);
    if ((t & 31) == 0) red[t >> 5] = sum;
    __syncthreads();
    sum = 0.0f;
    for (int i = 0; i < 8; ++i) sum += red[i];
    float inv = 1.0f / sum;

    for (int m = t; m < Lk; m += 256) {
        float v = s[m] * scale + ((causal && m > l) ? -1e9f : 0.0f);
        p[m] = (bf16)(__expf(v - mx) * inv);
    }
}

// ---------------- fused residual add + LayerNorm (D=512), f32 + bf16 out ----------------
__global__ __launch_bounds__(256) void ln_kernel(const float* __restrict__ A,
                                                 const float* __restrict__ Bc,
                                                 const float* __restrict__ g,
                                                 const float* __restrict__ be,
                                                 float* __restrict__ outF,
                                                 bf16* __restrict__ outB) {
    long long row = blockIdx.x;
    const float* a = A + row * 512;
    const float* b = Bc + row * 512;
    int t = threadIdx.x;
    float x0 = a[t] + b[t];
    float x1 = a[t + 256] + b[t + 256];
    float s = x0 + x1, q = x0 * x0 + x1 * x1;
    __shared__ float r1[8], r2[8];
    for (int o = 16; o; o >>= 1) { s += __shfl_xor(s, o, 32); q += __shfl_xor(q, o, 32); }
    if ((t & 31) == 0) { r1[t >> 5] = s; r2[t >> 5] = q; }
    __syncthreads();
    s = 0.0f; q = 0.0f;
    for (int i = 0; i < 8; ++i) { s += r1[i]; q += r2[i]; }
    float mean = s * (1.0f / 512.0f);
    float var  = q * (1.0f / 512.0f) - mean * mean;
    float inv  = rsqrtf(var + 1e-5f);
    float y0 = (x0 - mean) * inv * g[t] + be[t];
    float y1 = (x1 - mean) * inv * g[t + 256] + be[t + 256];
    outF[row * 512 + t] = y0;        outF[row * 512 + t + 256] = y1;
    outB[row * 512 + t] = (bf16)y0;  outB[row * 512 + t + 256] = (bf16)y1;
}

// ---------------- VAE reparameterization ----------------
__global__ __launch_bounds__(256) void latent_kernel(const float* __restrict__ mu,
                                                     const float* __restrict__ lv,
                                                     const float* __restrict__ eps,
                                                     float* __restrict__ lat, int n) {
    int i = blockIdx.x * 256 + threadIdx.x;
    if (i < n) lat[i] = eps[i] * __expf(0.5f * lv[i]) + mu[i];
}

// ---------------- seg_cat: bar-scatter latent + attribute embeddings (bf16) ----------------
__global__ __launch_bounds__(192) void segcat_kernel(const float* __restrict__ lat,
                                                     const int* __restrict__ bp,
                                                     const int* __restrict__ rcls,
                                                     const int* __restrict__ pcls,
                                                     const float* __restrict__ rw,
                                                     const float* __restrict__ pw,
                                                     bf16* __restrict__ out) {
    int t = blockIdx.x;          // t = tt*B + b
    int tt = t >> 2, b = t & 3;
    int j = threadIdx.x;         // 0..191
    __shared__ int sbar, svalid;
    if (j == 0) {
        int bar = 0;
        for (int i = 1; i <= 16; ++i) bar += (tt >= bp[b * 17 + i]) ? 1 : 0;
        sbar = bar < 15 ? bar : 15;
        svalid = (tt >= bp[b * 17]) && (tt < bp[b * 17 + 16]);
    }
    __syncthreads();
    const float attr_scale = 5.656854249492380f;   // sqrt(32)
    float v;
    if (j < 128)       v = svalid ? lat[(b * 16 + sbar) * 128 + j] : 0.0f;
    else if (j < 160)  v = rw[rcls[t] * 32 + (j - 128)] * attr_scale;
    else               v = pw[pcls[t] * 32 + (j - 160)] * attr_scale;
    out[(long long)t * 192 + j] = (bf16)v;
}

// ---------------- xl = x + seg_proj (f32 + bf16 out) ----------------
__global__ __launch_bounds__(256) void addcvt_kernel(const float* __restrict__ a,
                                                     const float* __restrict__ b,
                                                     float* __restrict__ of,
                                                     bf16* __restrict__ ob, long long n) {
    long long i = (long long)blockIdx.x * 256 + threadIdx.x;
    if (i < n) { float v = a[i] + b[i]; of[i] = v; ob[i] = (bf16)v; }
}

// =====================================================================
extern "C" void kernel_launch(void* const* d_in, const int* in_sizes, int n_in,
                              void* d_out, int out_size, void* d_ws, size_t ws_size,
                              hipStream_t stream) {
    const int*   enc_inp   = (const int*)d_in[0];
    const int*   dec_inp   = (const int*)d_in[1];
    const int*   bar_pos   = (const int*)d_in[2];
    const int*   rfreq_cls = (const int*)d_in[3];
    const int*   polyph_cls= (const int*)d_in[4];
    const float* eps       = (const float*)d_in[5];
    const float* tok_emb   = (const float*)d_in[6];
    const float* fc_mu_w   = (const float*)d_in[7];
    const float* fc_mu_b   = (const float*)d_in[8];
    const float* fc_lv_w   = (const float*)d_in[9];
    const float* fc_lv_b   = (const float*)d_in[10];
    const float* seg_proj_w= (const float*)d_in[11];
    const float* out_w     = (const float*)d_in[12];
    const float* out_b     = (const float*)d_in[13];
    const float* rfreq_w   = (const float*)d_in[14];
    const float* polyph_w  = (const float*)d_in[15];
    const float* W[24];
    for (int i = 0; i < 24; ++i) W[i] = (const float*)d_in[16 + i];
    float* out = (float*)d_out;

    // ---- workspace bump allocator ----
    char* p = (char*)d_ws;
    auto alloc = [&](size_t bytes) -> char* {
        char* r = p; p += (bytes + 255) & ~(size_t)255; return r;
    };
    bf16*  wq_b   = (bf16*) alloc(1536ll * 512 * 2);
    bf16*  wo_b   = (bf16*) alloc(512ll * 512 * 2);
    bf16*  wf1_b  = (bf16*) alloc(2048ll * 512 * 2);
    bf16*  wf2_b  = (bf16*) alloc(512ll * 2048 * 2);
    bf16*  wmu_b  = (bf16*) alloc(128ll * 512 * 2);
    bf16*  wlv_b  = (bf16*) alloc(128ll * 512 * 2);
    bf16*  wsp_b  = (bf16*) alloc(512ll * 192 * 2);
    bf16*  wout_b = (bf16*) alloc(330ll * 512 * 2);
    float* x_f    = (float*)alloc(8192ll * 512 * 4);
    bf16*  x_b    = (bf16*) alloc(8192ll * 512 * 2);
    float* xl_f   = (float*)alloc(4096ll * 512 * 4);
    bf16*  xl_b   = (bf16*) alloc(4096ll * 512 * 2);
    bf16*  qkv_b  = (bf16*) alloc(8192ll * 1536 * 2);
    float* sc_f   = (float*)alloc(33554432ll * 4);   // max(512*128*128, 32*1024*1024)
    bf16*  pr_b   = (bf16*) alloc(33554432ll * 2);
    bf16*  at_b   = (bf16*) alloc(8192ll * 512 * 2);
    float* op_f   = (float*)alloc(8192ll * 512 * 4);
    float* l1_f   = (float*)alloc(8192ll * 512 * 4);
    bf16*  l1_b   = (bf16*) alloc(8192ll * 512 * 2);
    bf16*  f1_b   = (bf16*) alloc(8192ll * 2048 * 2);
    float* f2_f   = (float*)alloc(8192ll * 512 * 4);
    float* lat    = (float*)alloc(64ll * 128 * 4);
    bf16*  sg_b   = (bf16*) alloc(4096ll * 192 * 2);
    float* sp_f   = (float*)alloc(4096ll * 512 * 4);

    auto cvt = [&](const float* src, bf16* dst, long long n) {
        long long chunks = (n + 3) / 4;
        cvt_kernel<<<(unsigned)((chunks + 255) / 256), 256, 0, stream>>>(src, dst, n);
    };
    auto gemm = [&](const bf16* A, long long aS1, long long aS2, int lda,
                    const bf16* B, long long bS1, long long bS2, int ldb, int transB,
                    float* Cf, bf16* Cb, long long cS1, long long cS2, int ldc,
                    const float* bias, int relu, int M, int N, int K, int Z, int nz2) {
        int aligned = (M % TM == 0) && (N % TN == 0) && (K % TK == 0);
        dim3 grid((N + TN - 1) / TN, (M + TM - 1) / TM, Z);
        if (aligned && transB)
            gemm_bf16_kernel<1, 1><<<grid, 256, 0, stream>>>(A, aS1, aS2, lda, B, bS1, bS2, ldb,
                Cf, Cb, cS1, cS2, ldc, bias, relu, M, N, K, nz2);
        else if (aligned && !transB)
            gemm_bf16_kernel<1, 0><<<grid, 256, 0, stream>>>(A, aS1, aS2, lda, B, bS1, bS2, ldb,
                Cf, Cb, cS1, cS2, ldc, bias, relu, M, N, K, nz2);
        else if (!aligned && transB)
            gemm_bf16_kernel<0, 1><<<grid, 256, 0, stream>>>(A, aS1, aS2, lda, B, bS1, bS2, ldb,
                Cf, Cb, cS1, cS2, ldc, bias, relu, M, N, K, nz2);
        else
            gemm_bf16_kernel<0, 0><<<grid, 256, 0, stream>>>(A, aS1, aS2, lda, B, bS1, bS2, ldb,
                Cf, Cb, cS1, cS2, ldc, bias, relu, M, N, K, nz2);
    };

    const float escale = sqrtf(512.0f);
    const float iscl   = 0.125f;                 // 1/sqrt(dh=64)

    // small weights -> bf16 (once)
    cvt(fc_mu_w, wmu_b, 128ll * 512);
    cvt(fc_lv_w, wlv_b, 128ll * 512);
    cvt(seg_proj_w, wsp_b, 512ll * 192);
    cvt(out_w, wout_b, 330ll * 512);

    // =========== ENCODER: 8192 tokens (row = l*64 + n), NB = 64 ===========
    embed_kernel<<<8192, 256, 0, stream>>>(tok_emb, enc_inp, x_f, x_b, 64, escale);
    for (int L = 0; L < 12; ++L) {
        cvt(W[0] + (long long)L * 1536 * 512, wq_b, 1536ll * 512);
        cvt(W[2] + (long long)L * 512 * 512, wo_b, 512ll * 512);
        cvt(W[4] + (long long)L * 2048 * 512, wf1_b, 2048ll * 512);
        cvt(W[6] + (long long)L * 512 * 2048, wf2_b, 512ll * 2048);
        gemm(x_b, 0, 0, 512, wq_b, 0, 0, 512, 1, nullptr, qkv_b, 0, 0, 1536,
             W[1] + L * 1536, 0, 8192, 1536, 512, 1, 1);
        gemm(qkv_b, 1536, 64, 64 * 1536, qkv_b + 512, 1536, 64, 64 * 1536, 1,
             sc_f, nullptr, 8ll * 16384, 16384, 128, nullptr, 0, 128, 128, 64, 512, 8);
        softmax_kernel<<<512 * 128, 256, 0, stream>>>(sc_f, pr_b, 128, 128, iscl, 0);
        gemm(pr_b, 8ll * 16384, 16384, 128, qkv_b + 1024, 1536, 64, 64 * 1536, 0,
             nullptr, at_b, 512, 64, 64 * 512, nullptr, 0, 128, 64, 128, 512, 8);
        gemm(at_b, 0, 0, 512, wo_b, 0, 0, 512, 1, op_f, nullptr, 0, 0, 512,
             W[3] + L * 512, 0, 8192, 512, 512, 1, 1);
        ln_kernel<<<8192, 256, 0, stream>>>(x_f, op_f, W[8] + L * 512, W[9] + L * 512, l1_f, l1_b);
        gemm(l1_b, 0, 0, 512, wf1_b, 0, 0, 512, 1, nullptr, f1_b, 0, 0, 2048,
             W[5] + L * 2048, 1, 8192, 2048, 512, 1, 1);
        gemm(f1_b, 0, 0, 2048, wf2_b, 0, 0, 2048, 1, f2_f, nullptr, 0, 0, 512,
             W[7] + L * 512, 0, 8192, 512, 2048, 1, 1);
        ln_kernel<<<8192, 256, 0, stream>>>(l1_f, f2_f, W[10] + L * 512, W[11] + L * 512, x_f, x_b);
    }
    // h0 = rows 0..63 (l=0) -> mu, logvar straight into d_out
    gemm(x_b, 0, 0, 512, wmu_b, 0, 0, 512, 1, out, nullptr, 0, 0, 128,
         fc_mu_b, 0, 64, 128, 512, 1, 1);
    gemm(x_b, 0, 0, 512, wlv_b, 0, 0, 512, 1, out + 8192, nullptr, 0, 0, 128,
         fc_lv_b, 0, 64, 128, 512, 1, 1);
    latent_kernel<<<32, 256, 0, stream>>>(out, out + 8192, eps, lat, 8192);

    // =========== segment conditioning ===========
    segcat_kernel<<<4096, 192, 0, stream>>>(lat, bar_pos, rfreq_cls, polyph_cls,
                                            rfreq_w, polyph_w, sg_b);
    gemm(sg_b, 0, 0, 192, wsp_b, 0, 0, 192, 1, sp_f, nullptr, 0, 0, 512,
         nullptr, 0, 4096, 512, 192, 1, 1);

    // =========== DECODER: 4096 tokens (row = t*4 + b), NB = 4, causal ===========
    embed_kernel<<<4096, 256, 0, stream>>>(tok_emb, dec_inp, x_f, x_b, 4, escale);
    for (int L = 0; L < 12; ++L) {
        cvt(W[12] + (long long)L * 1536 * 512, wq_b, 1536ll * 512);
        cvt(W[14] + (long long)L * 512 * 512, wo_b, 512ll * 512);
        cvt(W[16] + (long long)L * 2048 * 512, wf1_b, 2048ll * 512);
        cvt(W[18] + (long long)L * 512 * 2048, wf2_b, 512ll * 2048);
        addcvt_kernel<<<(4096 * 512) / 256, 256, 0, stream>>>(x_f, sp_f, xl_f, xl_b, 4096ll * 512);
        gemm(xl_b, 0, 0, 512, wq_b, 0, 0, 512, 1, nullptr, qkv_b, 0, 0, 1536,
             W[13] + L * 1536, 0, 4096, 1536, 512, 1, 1);
        gemm(qkv_b, 1536, 64, 4 * 1536, qkv_b + 512, 1536, 64, 4 * 1536, 1,
             sc_f, nullptr, 8ll * 1048576, 1048576, 1024, nullptr, 0, 1024, 1024, 64, 32, 8);
        softmax_kernel<<<32 * 1024, 256, 0, stream>>>(sc_f, pr_b, 1024, 1024, iscl, 1);
        gemm(pr_b, 8ll * 1048576, 1048576, 1024, qkv_b + 1024, 1536, 64, 4 * 1536, 0,
             nullptr, at_b, 512, 64, 4 * 512, nullptr, 0, 1024, 64, 1024, 32, 8);
        gemm(at_b, 0, 0, 512, wo_b, 0, 0, 512, 1, op_f, nullptr, 0, 0, 512,
             W[15] + L * 512, 0, 4096, 512, 512, 1, 1);
        ln_kernel<<<4096, 256, 0, stream>>>(xl_f, op_f, W[20] + L * 512, W[21] + L * 512, l1_f, l1_b);
        gemm(l1_b, 0, 0, 512, wf1_b, 0, 0, 512, 1, nullptr, f1_b, 0, 0, 2048,
             W[17] + L * 2048, 1, 4096, 2048, 512, 1, 1);
        gemm(f1_b, 0, 0, 2048, wf2_b, 0, 0, 2048, 1, f2_f, nullptr, 0, 0, 512,
             W[19] + L * 512, 0, 4096, 512, 2048, 1, 1);
        ln_kernel<<<4096, 256, 0, stream>>>(l1_f, f2_f, W[22] + L * 512, W[23] + L * 512, x_f, x_b);
    }
    // logits = out @ out_w^T + out_b  -> d_out + 16384  [4096, 330]
    gemm(x_b, 0, 0, 512, wout_b, 0, 0, 512, 1, out + 16384, nullptr, 0, 0, 330,
         out_b, 0, 4096, 330, 512, 1, 1);
}
